// UniCrystalFormerLayer_74268574482995
// MI455X (gfx1250) — matrix operand order repaired and checked by reading, not compiled, not run on hardware
//
#include <hip/hip_runtime.h>
#include <hip/hip_bf16.h>

// ---------------------------------------------------------------------------
// Types / helpers
// ---------------------------------------------------------------------------
typedef __bf16 bf16_t;
typedef __attribute__((ext_vector_type(16))) __bf16 v16bf;
typedef __attribute__((ext_vector_type(8)))  __bf16 v8bf;
typedef __attribute__((ext_vector_type(8)))  float  v8f;

#define DEV __device__ __forceinline__

static constexpr int NN = 10000;
static constexpr int EE = 120000;

DEV bf16_t f2bf(float f) {
    union { float f; unsigned u; } a; a.f = f;
    unsigned r = (a.u + 0x7FFFu + ((a.u >> 16) & 1u)) >> 16;
    union { unsigned short s; bf16_t b; } c; c.s = (unsigned short)r;
    return c.b;
}
DEV float siluf(float x) { return x / (1.f + __expf(-x)); }
DEV float sigf(float x)  { return 1.f / (1.f + __expf(-x)); }

// Async 16B copy global -> LDS (ASYNCcnt tracked, bypasses VGPRs).
DEV void async_copy16(unsigned lds_off, const bf16_t* gptr) {
    asm volatile("global_load_async_to_lds_b128 %0, %1, off"
                 :: "v"(lds_off), "v"(gptr) : "memory");
}
DEV void async_wait0() {
    asm volatile("s_wait_asynccnt 0x0" ::: "memory");
}
DEV unsigned lds_off32(const void* p) {
    return (unsigned)(unsigned long long)p;  // flat LDS aperture: addr[31:0]
}

// A fragment: 16x32 bf16 tile rows in LDS (row-major, ldk elems).
// Per ISA: lanes 0-15 row=lane, K = {0..7}+8*kh and {16..23}+8*kh contiguous.
DEV v16bf load_a_frag(const bf16_t* lds, int ldk, int kbase) {
    int lane = threadIdx.x & 31;
    int row = lane & 15, kh = lane >> 4;
    const bf16_t* p = lds + row * ldk + kbase + 8 * kh;
    union { v16bf v; v8bf h[2]; } u;
    u.h[0] = *(const v8bf*)(p);
    u.h[1] = *(const v8bf*)(p + 16);
    return u.v;
}

// B fragment: weight stored transposed Wt[Nout][K] bf16, so for lane col n the
// K-values are {0..7}+8*kh and {16..23}+8*kh (matching A).
DEV v16bf load_b_frag(const bf16_t* wt, int K, int nbase, int kbase) {
    int lane = threadIdx.x & 31;
    int col = lane & 15, kh = lane >> 4;
    const bf16_t* p = wt + (size_t)(nbase + col) * K + kbase + 8 * kh;
    union { v16bf v; v8bf h[2]; } u;
    u.h[0] = *(const v8bf*)(p);
    u.h[1] = *(const v8bf*)(p + 16);
    return u.v;
}

// 16x16 f32 tile = (16 x K) A_lds  @  (K x 16) column slice of Wt
DEV v8f wave_gemm(const bf16_t* ldsA, int ldk, const bf16_t* wt, int K, int nbase) {
    v8f acc = {0.f, 0.f, 0.f, 0.f, 0.f, 0.f, 0.f, 0.f};
    for (int kb = 0; kb < K; kb += 32) {
        v16bf a = load_a_frag(ldsA, ldk, kb);
        v16bf b = load_b_frag(wt, K, nbase, kb);
        acc = __builtin_amdgcn_wmma_f32_16x16x32_bf16(false, a, false, b,
                                                      (short)0, acc, false, false);
    }
    return acc;
}

// ---------------------------------------------------------------------------
// Elementwise f32 -> bf16 conversion (one-time; halves later gather traffic)
// ---------------------------------------------------------------------------
__global__ void k_cvt_bf16(const float* __restrict__ X, bf16_t* __restrict__ Y,
                           long n) {
    long i = (long)blockIdx.x * 256 + threadIdx.x;
    if (i < n) Y[i] = f2bf(X[i]);
}

// Weight convert+transpose: W (K x N) f32  ->  Wt (N x K) bf16
__global__ void k_wt(const float* __restrict__ W, bf16_t* __restrict__ Wt,
                     int K, int Nn) {
    int i = blockIdx.x * 256 + threadIdx.x;
    if (i >= K * Nn) return;
    int n = i / K, k = i % K;
    Wt[i] = f2bf(W[(size_t)k * Nn + n]);
}

// ---------------------------------------------------------------------------
// Generic row-tile GEMM:  Y(M x Nout) = act(A(M x K) @ W + bias)
// A is bf16 row-major; the 16-row tile is staged via async DMA to LDS.
// block = 128 threads (4 waves), blockIdx.x = 16-row tile. M % 16 == 0.
// ---------------------------------------------------------------------------
__global__ void k_gemm_rows(const bf16_t* __restrict__ A, int K,
                            const bf16_t* __restrict__ Wt,
                            const float* __restrict__ bias,
                            float* __restrict__ Y, int Nout, int act) {
    __shared__ __align__(16) bf16_t tile[16 * 384];
    int t = threadIdx.x;
    long rowbase = (long)blockIdx.x * 16;
    const bf16_t* Arow = A + rowbase * K;
    unsigned lbase = lds_off32(tile);
    int chunks = (16 * K) >> 3;                  // 16B chunks
    for (int ci = t; ci < chunks; ci += 128)
        async_copy16(lbase + (unsigned)(ci * 16), Arow + ci * 8);
    async_wait0();
    __syncthreads();
    int wave = t >> 5, lane = t & 31, col = lane & 15, kh = lane >> 4;
    for (int nt = wave; nt < (Nout >> 4); nt += 4) {
        v8f acc = wave_gemm(tile, K, Wt, K, nt * 16);
        for (int r = 0; r < 8; ++r) {
            int row = r + 8 * kh, c = nt * 16 + col;
            float x = acc[r] + bias[c];
            if (act == 1) x = siluf(x);
            Y[(rowbase + row) * (long)Nout + c] = x;
        }
    }
}

// ---------------------------------------------------------------------------
// CartNet edge pass 1: gate = MLPg(cat), msg = MLPa(cat); gate BN statistics.
// cat = [x[dst] | x[src] | edge_attr] gathered row-wise via async DMA to LDS.
// ---------------------------------------------------------------------------
__global__ void k_cart_edges(const bf16_t* __restrict__ xcb,
                             const int* __restrict__ ei,
                             const bf16_t* __restrict__ eacb,
                             const bf16_t* Wg1t, const float* bg1,
                             const bf16_t* Wg2t, const float* bg2,
                             const bf16_t* Wa1t, const float* ba1,
                             const bf16_t* Wa2t, const float* ba2,
                             float* __restrict__ gate, float* __restrict__ msg,
                             float* __restrict__ ssum_g, float* __restrict__ ssq_g) {
    __shared__ __align__(16) bf16_t cat[16 * 384];
    __shared__ __align__(16) bf16_t hid[16 * 128];
    __shared__ float ssum[128], ssq[128];
    __shared__ int dss[32];
    int t = threadIdx.x;
    long e0 = (long)blockIdx.x * 16;
    if (t < 16) { dss[t] = ei[EE + e0 + t]; dss[16 + t] = ei[e0 + t]; }
    ssum[t] = 0.f; ssq[t] = 0.f;
    __syncthreads();
    // gather 16 x (3x128) bf16 rows: 768 16-byte chunks via async DMA
    unsigned lbase = lds_off32(cat);
    for (int ci = t; ci < 768; ci += 128) {
        int r = ci / 48, s16 = ci % 48;
        int seg = s16 >> 4, o8 = (s16 & 15) * 8;  // 0:x[dst] 1:x[src] 2:eattr
        const bf16_t* g;
        if (seg == 0)      g = xcb  + (long)dss[r] * 128 + o8;
        else if (seg == 1) g = xcb  + (long)dss[16 + r] * 128 + o8;
        else               g = eacb + (e0 + r) * 128 + o8;
        async_copy16(lbase + (unsigned)((r * 384 + seg * 128 + o8) * 2), g);
    }
    async_wait0();
    __syncthreads();
    int wave = t >> 5, lane = t & 31, col = lane & 15, kh = lane >> 4;
    // ---- gate path ----
    for (int nt = wave; nt < 8; nt += 4) {
        v8f acc = wave_gemm(cat, 384, Wg1t, 384, nt * 16);
        for (int r = 0; r < 8; ++r)
            hid[(r + 8 * kh) * 128 + nt * 16 + col] = f2bf(siluf(acc[r] + bg1[nt * 16 + col]));
    }
    __syncthreads();
    for (int nt = wave; nt < 8; nt += 4) {
        v8f acc = wave_gemm(hid, 128, Wg2t, 128, nt * 16);
        for (int r = 0; r < 8; ++r) {
            int row = r + 8 * kh, c = nt * 16 + col;
            float x = acc[r] + bg2[c];
            gate[(e0 + row) * 128 + c] = x;
            atomicAdd(&ssum[c], x);
            atomicAdd(&ssq[c], x * x);
        }
    }
    __syncthreads();
    atomicAdd(&ssum_g[t], ssum[t]);
    atomicAdd(&ssq_g[t], ssq[t]);
    __syncthreads();
    // ---- aggr path (reuse hid) ----
    for (int nt = wave; nt < 8; nt += 4) {
        v8f acc = wave_gemm(cat, 384, Wa1t, 384, nt * 16);
        for (int r = 0; r < 8; ++r)
            hid[(r + 8 * kh) * 128 + nt * 16 + col] = f2bf(siluf(acc[r] + ba1[nt * 16 + col]));
    }
    __syncthreads();
    for (int nt = wave; nt < 8; nt += 4) {
        v8f acc = wave_gemm(hid, 128, Wa2t, 128, nt * 16);
        for (int r = 0; r < 8; ++r) {
            int row = r + 8 * kh, c = nt * 16 + col;
            msg[(e0 + row) * 128 + c] = acc[r] + ba2[c];
        }
    }
}

// finalize per-channel BN: scale = g*rsqrt(var+eps), shift = b - mean*scale
__global__ void k_bn_fin(const float* sum, const float* sq,
                         const float* g, const float* b,
                         float rows, float* scale, float* shift) {
    int c = threadIdx.x;
    float mu = sum[c] / rows;
    float var = sq[c] / rows - mu * mu;
    float s = g[c] * rsqrtf(var + 1e-5f);
    scale[c] = s;
    shift[c] = b[c] - mu * s;
}

// CartNet pass 2: e_ij = sigmoid(BN(gate)) * envelope(d);  x_agg[dst] += e_ij*msg
__global__ void k_cart_scatter(const float* __restrict__ gate,
                               const float* __restrict__ msg,
                               const float* __restrict__ dist,
                               const int* __restrict__ ei,
                               const float* __restrict__ scale,
                               const float* __restrict__ shift,
                               float* __restrict__ x_agg) {
    long i = (long)blockIdx.x * 256 + threadIdx.x;
    if (i >= (long)EE * 128) return;
    int e = (int)(i >> 7), c = (int)(i & 127);
    float s = sigf(scale[c] * gate[i] + shift[c]);
    float d = dist[e];
    float env = (d < 8.0f) ? 0.5f * (__cosf(0.39269908f * d) + 1.f) : 0.f;
    atomicAdd(&x_agg[(long)ei[EE + e] * 128 + c], s * env * msg[i]);
}

// column sums / sumsq over (rows x 128) f32
__global__ void k_colstats(const float* __restrict__ X, long rows,
                           float* __restrict__ sum, float* __restrict__ sq) {
    int c = threadIdx.x;
    float s = 0.f, q = 0.f;
    for (long r = blockIdx.x; r < rows; r += gridDim.x) {
        float v = X[r * 128 + c];
        s += v; q += v * v;
    }
    atomicAdd(&sum[c], s);
    atomicAdd(&sq[c], q);
}

// out = silu(BN(agg)) (+ optional residual)
__global__ void k_bnsilu(const float* __restrict__ agg,
                         const float* __restrict__ scale,
                         const float* __restrict__ shift,
                         const float* __restrict__ resid,
                         float* __restrict__ out, long total) {
    long i = (long)blockIdx.x * 256 + threadIdx.x;
    if (i >= total) return;
    int c = (int)(i & 127);
    float x = siluf(scale[c] * agg[i] + shift[c]);
    out[i] = resid ? (x + resid[i]) : x;
}

// ---------------------------------------------------------------------------
// Matformer edge kernel (per 16-edge tile, per head):
//  alpha = q3*k3/sqrt(3C) -> LN -> sigmoid   (elementwise, in LDS)
//  m = (concat(v_d,v_s,ee) @ Wmu + bmu) * sact -> @ Wml + bml -> LN -> scatter
// ---------------------------------------------------------------------------
__global__ void k_matformer(const float* __restrict__ q, const float* __restrict__ k,
                            const float* __restrict__ v, const float* __restrict__ ee,
                            const int* __restrict__ ei,
                            const bf16_t* Wmut, const float* bmu,
                            const bf16_t* Wmlt, const float* bml,
                            const float* ln3g, const float* ln3b,
                            const float* lnmg, const float* lnmb,
                            float* __restrict__ out_agg) {
    __shared__ __align__(16) bf16_t min_[16 * 384];
    __shared__ float sact[16 * 384];
    __shared__ __align__(16) bf16_t m2[16 * 384];
    __shared__ float zb[16 * 128];
    __shared__ float mu16[16], rs16[16];
    __shared__ int dss[32];
    int t = threadIdx.x;
    long e0 = (long)blockIdx.x * 16;
    int h = blockIdx.y;
    if (t < 16) { dss[t] = ei[EE + e0 + t]; dss[16 + t] = ei[e0 + t]; }
    __syncthreads();
    const float inv = 0.051031036307983f; // 1/sqrt(384)
    for (int i = t; i < 16 * 384; i += 128) {
        int r = i / 384, c = i % 384, cc = c & 127;
        int dn = dss[r], sn = dss[16 + r];
        float qv = q[(long)dn * 256 + h * 128 + cc];
        float kv, vv;
        if (c < 128)      { kv = k[(long)dn * 256 + h * 128 + cc]; vv = v[(long)dn * 256 + h * 128 + cc]; }
        else if (c < 256) { kv = k[(long)sn * 256 + h * 128 + cc]; vv = v[(long)sn * 256 + h * 128 + cc]; }
        else              { float evl = ee[(e0 + r) * 256 + h * 128 + cc]; kv = evl; vv = evl; }
        sact[i] = qv * kv * inv;
        min_[i] = f2bf(vv);
    }
    __syncthreads();
    if (t < 16) {
        float s = 0.f, q2 = 0.f;
        for (int c = 0; c < 384; ++c) { float a = sact[t * 384 + c]; s += a; q2 += a * a; }
        float mu = s * (1.f / 384.f);
        float var = q2 * (1.f / 384.f) - mu * mu;
        mu16[t] = mu; rs16[t] = rsqrtf(var + 1e-5f);
    }
    __syncthreads();
    for (int i = t; i < 16 * 384; i += 128) {
        int r = i / 384, c = i % 384;
        sact[i] = sigf(ln3g[c] * (sact[i] - mu16[r]) * rs16[r] + ln3b[c]);
    }
    __syncthreads();
    int wave = t >> 5, lane = t & 31, col = lane & 15, kh = lane >> 4;
    for (int nt = wave; nt < 24; nt += 4) {
        v8f acc = wave_gemm(min_, 384, Wmut, 384, nt * 16);
        for (int r = 0; r < 8; ++r) {
            int row = r + 8 * kh, c = nt * 16 + col;
            m2[row * 384 + c] = f2bf((acc[r] + bmu[c]) * sact[row * 384 + c]);
        }
    }
    __syncthreads();
    for (int nt = wave; nt < 8; nt += 4) {
        v8f acc = wave_gemm(m2, 384, Wmlt, 384, nt * 16);
        for (int r = 0; r < 8; ++r)
            zb[(r + 8 * kh) * 128 + nt * 16 + col] = acc[r] + bml[nt * 16 + col];
    }
    __syncthreads();
    if (t < 16) {
        float s = 0.f, q2 = 0.f;
        for (int c = 0; c < 128; ++c) { float a = zb[t * 128 + c]; s += a; q2 += a * a; }
        float mu = s * (1.f / 128.f);
        float var = q2 * (1.f / 128.f) - mu * mu;
        mu16[t] = mu; rs16[t] = rsqrtf(var + 1e-5f);
    }
    __syncthreads();
    for (int i = t; i < 16 * 128; i += 128) {
        int r = i >> 7, c = i & 127;
        float x = lnmg[c] * (zb[i] - mu16[r]) * rs16[r] + lnmb[c];
        atomicAdd(&out_agg[(long)dss[r] * 128 + c], 0.5f * x); // head mean (H=2)
    }
}

// beta gate + combine:  x_m = beta*x_r + (1-beta)*out
__global__ void k_beta(const float* __restrict__ outb, const float* __restrict__ x_r,
                       const float* __restrict__ Wb, float* __restrict__ x_m) {
    int n = blockIdx.x * 128 + threadIdx.x;
    if (n >= NN) return;
    float s = 0.f;
    for (int c = 0; c < 128; ++c) {
        float o = outb[(long)n * 128 + c], r = x_r[(long)n * 128 + c];
        s += o * Wb[c] + r * Wb[128 + c] + (o - r) * Wb[256 + c];
    }
    float beta = sigf(s);
    for (int c = 0; c < 128; ++c) {
        long i = (long)n * 128 + c;
        x_m[i] = beta * x_r[i] + (1.f - beta) * outb[i];
    }
}

// CrossMix gated fusion + final outputs (2,N,C)
__global__ void k_crossmix(const float* __restrict__ x_c, const float* __restrict__ x_m,
                           const bf16_t* Wclt, const float* bcl,
                           const bf16_t* Wcgt, const float* bcg,
                           float* __restrict__ out) {
    __shared__ __align__(16) bf16_t xc[16 * 256];
    __shared__ float gbuf[16 * 128];
    __shared__ float fbuf[16 * 128];
    int t = threadIdx.x;
    long n0 = (long)blockIdx.x * 16;
    for (int i = t; i < 16 * 256; i += 128) {
        int r = i >> 8, c = i & 255;
        float vv = (c < 128) ? x_c[(n0 + r) * 128 + c] : x_m[(n0 + r) * 128 + (c - 128)];
        xc[i] = f2bf(vv);
    }
    __syncthreads();
    int wave = t >> 5, lane = t & 31, col = lane & 15, kh = lane >> 4;
    for (int task = wave; task < 16; task += 4) {
        int nt = task & 7;
        bool isg = task >= 8;
        v8f acc = wave_gemm(xc, 256, isg ? Wcgt : Wclt, 256, nt * 16);
        for (int r = 0; r < 8; ++r) {
            int row = r + 8 * kh, c = nt * 16 + col;
            if (isg) gbuf[row * 128 + c] = sigf(acc[r] + bcg[c]);
            else     fbuf[row * 128 + c] = acc[r] + bcl[c];
        }
    }
    __syncthreads();
    for (int i = t; i < 16 * 128; i += 128) {
        int r = i >> 7, c = i & 127;
        long gi = (n0 + r) * 128 + c;
        float g = gbuf[i], f = fbuf[i];
        float xcv = x_c[gi], xmv = x_m[gi];
        float xo = g * f + (1.f - g) * (xcv + xmv) * 0.5f;
        out[gi] = xo + xcv;
        out[(long)NN * 128 + gi] = xo + xmv;
    }
}

// ---------------------------------------------------------------------------
// Host launcher
// ---------------------------------------------------------------------------
extern "C" void kernel_launch(void* const* d_in, const int* in_sizes, int n_in,
                              void* d_out, int out_size, void* d_ws, size_t ws_size,
                              hipStream_t stream) {
    (void)in_sizes; (void)n_in; (void)out_size; (void)ws_size;

    const float* x_cart = (const float*)d_in[0];
    const int*   ei_c   = (const int*)d_in[1];
    const float* ea_c   = (const float*)d_in[2];
    const float* dist   = (const float*)d_in[3];
    const float* x_mat  = (const float*)d_in[4];
    const int*   ei_m   = (const int*)d_in[5];
    const float* ea_m   = (const float*)d_in[6];
    int p = 7;
    const float* Wg1 = (const float*)d_in[p + 0];
    const float* bg1 = (const float*)d_in[p + 1];
    const float* Wg2 = (const float*)d_in[p + 2];
    const float* bg2 = (const float*)d_in[p + 3];
    const float* Wa1 = (const float*)d_in[p + 4];
    const float* ba1 = (const float*)d_in[p + 5];
    const float* Wa2 = (const float*)d_in[p + 6];
    const float* ba2 = (const float*)d_in[p + 7];
    const float* bne_g = (const float*)d_in[p + 8];
    const float* bne_b = (const float*)d_in[p + 9];
    const float* bnn_g = (const float*)d_in[p + 10];
    const float* bnn_b = (const float*)d_in[p + 11];
    const float* Wq = (const float*)d_in[p + 12];
    const float* bq = (const float*)d_in[p + 13];
    const float* Wk = (const float*)d_in[p + 14];
    const float* bk = (const float*)d_in[p + 15];
    const float* Wv = (const float*)d_in[p + 16];
    const float* bv = (const float*)d_in[p + 17];
    const float* We = (const float*)d_in[p + 18];
    const float* be = (const float*)d_in[p + 19];
    const float* Wmu = (const float*)d_in[p + 20];
    const float* bmu = (const float*)d_in[p + 21];
    const float* Wml = (const float*)d_in[p + 22];
    const float* bml = (const float*)d_in[p + 23];
    const float* ln3g = (const float*)d_in[p + 24];
    const float* ln3b = (const float*)d_in[p + 25];
    const float* lnmg = (const float*)d_in[p + 26];
    const float* lnmb = (const float*)d_in[p + 27];
    const float* bnm_g = (const float*)d_in[p + 28];
    const float* bnm_b = (const float*)d_in[p + 29];
    const float* Ws = (const float*)d_in[p + 30];
    const float* bs = (const float*)d_in[p + 31];
    const float* Wb = (const float*)d_in[p + 32];
    const float* Wcl = (const float*)d_in[p + 33];
    const float* bcl = (const float*)d_in[p + 34];
    const float* Wcg = (const float*)d_in[p + 35];
    const float* bcg = (const float*)d_in[p + 36];

    // workspace bump allocator
    char* wsp = (char*)d_ws;
    size_t off = 0;
    auto alloc = [&](size_t bytes) -> void* {
        void* ptr = wsp + off;
        off += (bytes + 255) & ~(size_t)255;
        return ptr;
    };
    // bf16 transposed weights
    bf16_t* Wg1t = (bf16_t*)alloc(128 * 384 * 2);
    bf16_t* Wg2t = (bf16_t*)alloc(128 * 128 * 2);
    bf16_t* Wa1t = (bf16_t*)alloc(128 * 384 * 2);
    bf16_t* Wa2t = (bf16_t*)alloc(128 * 128 * 2);
    bf16_t* Wqt  = (bf16_t*)alloc(256 * 128 * 2);
    bf16_t* Wkt  = (bf16_t*)alloc(256 * 128 * 2);
    bf16_t* Wvt  = (bf16_t*)alloc(256 * 128 * 2);
    bf16_t* Weet = (bf16_t*)alloc(256 * 128 * 2);
    bf16_t* Wmut = (bf16_t*)alloc(384 * 384 * 2);
    bf16_t* Wmlt = (bf16_t*)alloc(128 * 384 * 2);
    bf16_t* Wst  = (bf16_t*)alloc(128 * 128 * 2);
    bf16_t* Wclt = (bf16_t*)alloc(128 * 256 * 2);
    bf16_t* Wcgt = (bf16_t*)alloc(128 * 256 * 2);
    // bf16 feature copies (async-DMA friendly)
    bf16_t* xcb  = (bf16_t*)alloc((size_t)NN * 128 * 2);
    bf16_t* xmb  = (bf16_t*)alloc((size_t)NN * 128 * 2);
    bf16_t* eacb = (bf16_t*)alloc((size_t)EE * 128 * 2);
    bf16_t* eamb = (bf16_t*)alloc((size_t)EE * 128 * 2);
    // f32 intermediates
    float* qf   = (float*)alloc((size_t)NN * 256 * 4);
    float* kf   = (float*)alloc((size_t)NN * 256 * 4);
    float* vf   = (float*)alloc((size_t)NN * 256 * 4);
    float* eef  = (float*)alloc((size_t)EE * 256 * 4);
    float* gate = (float*)alloc((size_t)EE * 128 * 4);
    float* msg  = (float*)alloc((size_t)EE * 128 * 4);
    float* x_agg   = (float*)alloc((size_t)NN * 128 * 4);
    float* out_agg = (float*)alloc((size_t)NN * 128 * 4);
    float* x_c  = (float*)alloc((size_t)NN * 128 * 4);
    float* x_r  = (float*)alloc((size_t)NN * 128 * 4);
    float* x_m  = (float*)alloc((size_t)NN * 128 * 4);
    float* outb = (float*)alloc((size_t)NN * 128 * 4);
    float* stats = (float*)alloc(12 * 128 * 4); // 3x(sum,sq) + 3x(scale,shift)
    float* sum_e = stats + 0 * 128,  *sq_e = stats + 1 * 128;
    float* sum_n = stats + 2 * 128,  *sq_n = stats + 3 * 128;
    float* sum_m = stats + 4 * 128,  *sq_m = stats + 5 * 128;
    float* sc_e = stats + 6 * 128,   *sh_e = stats + 7 * 128;
    float* sc_n = stats + 8 * 128,   *sh_n = stats + 9 * 128;
    float* sc_m = stats + 10 * 128,  *sh_m = stats + 11 * 128;

    hipMemsetAsync(x_agg, 0, (size_t)NN * 128 * 4, stream);
    hipMemsetAsync(out_agg, 0, (size_t)NN * 128 * 4, stream);
    hipMemsetAsync(stats, 0, 6 * 128 * 4, stream);

    auto xp = [&](const float* W, bf16_t* Wt, int K, int Nn_) {
        int tot = K * Nn_;
        k_wt<<<dim3((tot + 255) / 256), dim3(256), 0, stream>>>(W, Wt, K, Nn_);
    };
    xp(Wg1, Wg1t, 384, 128);  xp(Wg2, Wg2t, 128, 128);
    xp(Wa1, Wa1t, 384, 128);  xp(Wa2, Wa2t, 128, 128);
    xp(Wq,  Wqt,  128, 256);  xp(Wk,  Wkt,  128, 256);
    xp(Wv,  Wvt,  128, 256);  xp(We,  Weet, 128, 256);
    xp(Wmu, Wmut, 384, 384);  xp(Wml, Wmlt, 384, 128);
    xp(Ws,  Wst,  128, 128);  xp(Wcl, Wclt, 256, 128);
    xp(Wcg, Wcgt, 256, 128);

    auto cvt = [&](const float* X, bf16_t* Y, long n) {
        k_cvt_bf16<<<dim3((unsigned)((n + 255) / 256)), dim3(256), 0, stream>>>(X, Y, n);
    };
    cvt(x_cart, xcb, (long)NN * 128);
    cvt(x_mat,  xmb, (long)NN * 128);
    cvt(ea_c,   eacb, (long)EE * 128);
    cvt(ea_m,   eamb, (long)EE * 128);

    dim3 blk(128);
    // node GEMMs: q,k,v, skip (x_r), edge embed (ee)
    k_gemm_rows<<<dim3(NN / 16), blk, 0, stream>>>(xmb, 128, Wqt, bq, qf, 256, 0);
    k_gemm_rows<<<dim3(NN / 16), blk, 0, stream>>>(xmb, 128, Wkt, bk, kf, 256, 0);
    k_gemm_rows<<<dim3(NN / 16), blk, 0, stream>>>(xmb, 128, Wvt, bv, vf, 256, 0);
    k_gemm_rows<<<dim3(NN / 16), blk, 0, stream>>>(xmb, 128, Wst, bs, x_r, 128, 0);
    k_gemm_rows<<<dim3(EE / 16), blk, 0, stream>>>(eamb, 128, Weet, be, eef, 256, 0);

    // CartNet edge MLPs (+ gate batch stats)
    k_cart_edges<<<dim3(EE / 16), blk, 0, stream>>>(xcb, ei_c, eacb,
        Wg1t, bg1, Wg2t, bg2, Wa1t, ba1, Wa2t, ba2, gate, msg, sum_e, sq_e);
    k_bn_fin<<<dim3(1), dim3(128), 0, stream>>>(sum_e, sq_e, bne_g, bne_b, (float)EE, sc_e, sh_e);
    k_cart_scatter<<<dim3((EE * 128 + 255) / 256), dim3(256), 0, stream>>>(
        gate, msg, dist, ei_c, sc_e, sh_e, x_agg);
    // node BN + silu + residual -> x_c
    k_colstats<<<dim3(120), dim3(128), 0, stream>>>(x_agg, NN, sum_n, sq_n);
    k_bn_fin<<<dim3(1), dim3(128), 0, stream>>>(sum_n, sq_n, bnn_g, bnn_b, (float)NN, sc_n, sh_n);
    k_bnsilu<<<dim3((NN * 128 + 255) / 256), dim3(256), 0, stream>>>(
        x_agg, sc_n, sh_n, x_cart, x_c, (long)NN * 128);

    // Matformer edges (per head)
    k_matformer<<<dim3(EE / 16, 2), blk, 0, stream>>>(qf, kf, vf, eef, ei_m,
        Wmut, bmu, Wmlt, bml, ln3g, ln3b, lnmg, lnmb, out_agg);
    // out BN + silu
    k_colstats<<<dim3(120), dim3(128), 0, stream>>>(out_agg, NN, sum_m, sq_m);
    k_bn_fin<<<dim3(1), dim3(128), 0, stream>>>(sum_m, sq_m, bnm_g, bnm_b, (float)NN, sc_m, sh_m);
    k_bnsilu<<<dim3((NN * 128 + 255) / 256), dim3(256), 0, stream>>>(
        out_agg, sc_m, sh_m, (const float*)nullptr, outb, (long)NN * 128);
    // beta gate -> x_m
    k_beta<<<dim3((NN + 127) / 128), dim3(128), 0, stream>>>(outb, x_r, Wb, x_m);

    // CrossMix + final outputs
    k_crossmix<<<dim3(NN / 16), blk, 0, stream>>>(x_c, x_m, Wclt, bcl, Wcgt, bcg,
                                                  (float*)d_out);
}
